// GCN2_24592982737082
// MI455X (gfx1250) — compile-verified, matrix operand order checked by reading, not run on hardware
//
#include <hip/hip_runtime.h>
#include <hip/hip_bf16.h>

// ---------------------------------------------------------------------------
// GCN: 5x GraphConv(D=512) + residual + relu, mean-pool over 64 graphs,
// final [64,512]@[512,128] projection.
// GEMMs run on v_wmma_f32_16x16x32_f16 (f16 in, f32 accumulate).
// Weights pre-converted once per launch to transposed f16 (L2-resident).
// ---------------------------------------------------------------------------

#define N_NODES 10000
#define N_EDGES 160000
#define DIM     512
#define OUTC    128
#define NGRAPH  64
#define NPAD    10112          // 79 * 128
#define LDSW    40             // padded halves per LDS tile row (80B, 16B-aligned)
#define KSTEPS  (DIM / 32)     // 16

typedef float     v8f  __attribute__((ext_vector_type(8)));
typedef _Float16  h4v  __attribute__((ext_vector_type(4)));
typedef _Float16  h8v  __attribute__((ext_vector_type(8)));
typedef _Float16  h16v __attribute__((ext_vector_type(16)));

// ---------------------------------------------------------------------------
// utility kernels
// ---------------------------------------------------------------------------
__global__ void gcn_zero_f32(float* __restrict__ p, int n4) {
  int t = blockIdx.x * blockDim.x + threadIdx.x;
  if (t < n4) ((float4*)p)[t] = make_float4(0.f, 0.f, 0.f, 0.f);
}

__global__ void gcn_copy_pad(const float* __restrict__ src, float* __restrict__ dst,
                             int ncopy4, int ntot4) {
  int t = blockIdx.x * blockDim.x + threadIdx.x;
  if (t < ntot4) {
    if (t < ncopy4) ((float4*)dst)[t] = ((const float4*)src)[t];
    else            ((float4*)dst)[t] = make_float4(0.f, 0.f, 0.f, 0.f);
  }
}

// W [512,512] f32 row-major  ->  WT [512,512] f16, WT[n][k] = W[k][n]
__global__ void gcn_wt_convert(const float* __restrict__ W, _Float16* __restrict__ WT) {
  int t = blockIdx.x * blockDim.x + threadIdx.x;   // 512*512
  int k = t >> 9;
  int n = t & 511;
  WT[(size_t)n * DIM + k] = (_Float16)W[(size_t)k * DIM + n];
}

// ---------------------------------------------------------------------------
// scatter-add: agg[dst] += x[src] over all edges (8 channels / thread)
// ---------------------------------------------------------------------------
__global__ void gcn_scatter(const float* __restrict__ x,
                            const int* __restrict__ src,
                            const int* __restrict__ dst,
                            float* __restrict__ agg) {
  int t = blockIdx.x * blockDim.x + threadIdx.x;          // E * 64 threads
  int e = t >> 6;
  int c = (t & 63) << 3;
  if (e < N_EDGES) {
    int s = src[e];
    int d = dst[e];
    const float* xp = x + (size_t)s * DIM + c;
    float4 a = *(const float4*)(xp);
    float4 b = *(const float4*)(xp + 4);
    float* ap = agg + (size_t)d * DIM + c;
    atomicAdd(ap + 0, a.x); atomicAdd(ap + 1, a.y);
    atomicAdd(ap + 2, a.z); atomicAdd(ap + 3, a.w);
    atomicAdd(ap + 4, b.x); atomicAdd(ap + 5, b.y);
    atomicAdd(ap + 6, b.z); atomicAdd(ap + 7, b.w);
  }
}

// ---------------------------------------------------------------------------
// fragment load: two 16-byte ds_load_b128 per 16x32 f16 fragment
// ---------------------------------------------------------------------------
__device__ __forceinline__ h16v gcn_frag(const _Float16* s, int row, int h) {
  const h8v* p0 = (const h8v*)(s + row * LDSW + 8 * h);
  const h8v* p1 = (const h8v*)(s + row * LDSW + 16 + 8 * h);
  union { h16v v; h8v p[2]; } u;
  u.p[0] = *p0;
  u.p[1] = *p1;
  return u.v;
}

// ---------------------------------------------------------------------------
// fused layer GEMM:  xout = maybe_relu( xin + agg@Wr + b + xin@Ws )
// block = 256 thr (8 waves), tile = 128(M) x 128(N), K stepped by 32.
// Global loads for step kt+1 are issued before the WMMA phase of step kt
// (register-staged software pipeline over a single LDS buffer).
// ---------------------------------------------------------------------------
__global__ void __launch_bounds__(256)
gcn_layer_gemm(const float* __restrict__ xin,
               const float* __restrict__ agg,
               const _Float16* __restrict__ WrT,   // [n][k] f16
               const _Float16* __restrict__ WsT,   // [n][k] f16
               const float* __restrict__ bias,
               float* __restrict__ xout,
               int do_relu) {
  __shared__ _Float16 sAgg[128 * LDSW];
  __shared__ _Float16 sX  [128 * LDSW];
  __shared__ _Float16 sWr [128 * LDSW];
  __shared__ _Float16 sWs [128 * LDSW];

  const int m0 = blockIdx.x * 128;
  const int n0 = blockIdx.y * 128;
  const int tid  = threadIdx.x;
  const int lane = tid & 31;
  const int wv   = tid >> 5;
  const int l15  = lane & 15;
  const int hh   = lane >> 4;

  // ---- per-thread staging coordinates (fixed across K steps) --------------
  // A-type tiles: 128 rows x 8 float4-chunks, 4 units / thread
  int arow[4], achk[4];
#pragma unroll
  for (int u = 0; u < 4; ++u) {
    int idx = tid + 256 * u;
    arow[u] = idx >> 3;
    achk[u] = idx & 7;
  }
  // weight tiles (f16, pre-transposed): 128 rows x 4 h8v-chunks, 2 units/thr
  int wrow[2], wchk[2];
#pragma unroll
  for (int u = 0; u < 2; ++u) {
    int idx = tid + 256 * u;
    wrow[u] = idx >> 2;
    wchk[u] = idx & 3;
  }

  v8f acc[8];
#pragma unroll
  for (int f = 0; f < 8; ++f) acc[f] = v8f{};

  float4 va[4], vx[4];
  h8v    rwr[2], rws[2];

  // ---- prologue: issue loads for K-step 0 ---------------------------------
#pragma unroll
  for (int u = 0; u < 4; ++u) {
    size_t goff = (size_t)(m0 + arow[u]) * DIM + achk[u] * 4;
    va[u] = *(const float4*)(agg + goff);
    vx[u] = *(const float4*)(xin + goff);
  }
#pragma unroll
  for (int u = 0; u < 2; ++u) {
    size_t woff = (size_t)(n0 + wrow[u]) * DIM + wchk[u] * 8;
    rwr[u] = *(const h8v*)(WrT + woff);
    rws[u] = *(const h8v*)(WsT + woff);
  }

  for (int kt = 0; kt < KSTEPS; ++kt) {
    __syncthreads();                       // compute of kt-1 done with LDS

    // ---- store staged registers for step kt into LDS ----------------------
#pragma unroll
    for (int u = 0; u < 4; ++u) {
      h4v ha, hx;
      ha[0] = (_Float16)va[u].x; ha[1] = (_Float16)va[u].y;
      ha[2] = (_Float16)va[u].z; ha[3] = (_Float16)va[u].w;
      hx[0] = (_Float16)vx[u].x; hx[1] = (_Float16)vx[u].y;
      hx[2] = (_Float16)vx[u].z; hx[3] = (_Float16)vx[u].w;
      *(h4v*)(sAgg + arow[u] * LDSW + achk[u] * 4) = ha;
      *(h4v*)(sX   + arow[u] * LDSW + achk[u] * 4) = hx;
    }
#pragma unroll
    for (int u = 0; u < 2; ++u) {
      *(h8v*)(sWr + wrow[u] * LDSW + wchk[u] * 8) = rwr[u];
      *(h8v*)(sWs + wrow[u] * LDSW + wchk[u] * 8) = rws[u];
    }

    // ---- issue global loads for step kt+1 (overlap with WMMA phase) -------
    if (kt + 1 < KSTEPS) {
      const int k1 = (kt + 1) * 32;
#pragma unroll
      for (int u = 0; u < 4; ++u) {
        size_t goff = (size_t)(m0 + arow[u]) * DIM + k1 + achk[u] * 4;
        va[u] = *(const float4*)(agg + goff);
        vx[u] = *(const float4*)(xin + goff);
      }
#pragma unroll
      for (int u = 0; u < 2; ++u) {
        size_t woff = (size_t)(n0 + wrow[u]) * DIM + k1 + wchk[u] * 8;
        rwr[u] = *(const h8v*)(WrT + woff);
        rws[u] = *(const h8v*)(WsT + woff);
      }
    }

    __syncthreads();                       // staging for kt visible

    // ---- compute: each wave -> 16 rows x 128 cols -------------------------
    h16v aAgg = gcn_frag(sAgg, 16 * wv + l15, hh);
    h16v aX   = gcn_frag(sX,   16 * wv + l15, hh);
#pragma unroll
    for (int f = 0; f < 8; ++f) {
      h16v bR = gcn_frag(sWr, 16 * f + l15, hh);
      h16v bS = gcn_frag(sWs, 16 * f + l15, hh);
      acc[f] = __builtin_amdgcn_wmma_f32_16x16x32_f16(
          false, aAgg, false, bR, (short)0, acc[f], false, false);
      acc[f] = __builtin_amdgcn_wmma_f32_16x16x32_f16(
          false, aX, false, bS, (short)0, acc[f], false, false);
    }
  }

  // ---- epilogue: + bias + residual (+ relu), f32 out ----------------------
#pragma unroll
  for (int f = 0; f < 8; ++f) {
    int col = n0 + 16 * f + l15;
    float bv = bias[col];
#pragma unroll
    for (int r = 0; r < 8; ++r) {
      int row = m0 + 16 * wv + r + 8 * hh;
      size_t off = (size_t)row * DIM + col;
      float v = acc[f][r] + bv + xin[off];
      if (do_relu) v = fmaxf(v, 0.0f);
      xout[off] = v;
    }
  }
}

// ---------------------------------------------------------------------------
// mean-pool accumulation: pooled[g] += x[i], cnt[g] += 1
// ---------------------------------------------------------------------------
__global__ void gcn_pool(const float* __restrict__ x,
                         const int* __restrict__ batch,
                         float* __restrict__ pooled,
                         float* __restrict__ cnt) {
  int t = blockIdx.x * blockDim.x + threadIdx.x;      // N * 64 threads
  int i = t >> 6;
  int c = (t & 63) << 3;
  if (i < N_NODES) {
    int g = batch[i];
    const float* xp = x + (size_t)i * DIM + c;
    float4 a = *(const float4*)(xp);
    float4 b = *(const float4*)(xp + 4);
    float* pp = pooled + (size_t)g * DIM + c;
    atomicAdd(pp + 0, a.x); atomicAdd(pp + 1, a.y);
    atomicAdd(pp + 2, a.z); atomicAdd(pp + 3, a.w);
    atomicAdd(pp + 4, b.x); atomicAdd(pp + 5, b.y);
    atomicAdd(pp + 6, b.z); atomicAdd(pp + 7, b.w);
    if (c == 0) atomicAdd(cnt + g, 1.0f);
  }
}

// ---------------------------------------------------------------------------
// final projection: out[g,o] = mean(pooled[g])@Wlin + blin   (8 MFLOP)
// ---------------------------------------------------------------------------
__global__ void gcn_final(const float* __restrict__ pooled,
                          const float* __restrict__ cnt,
                          const float* __restrict__ Wlin,
                          const float* __restrict__ blin,
                          float* __restrict__ out) {
  int t = blockIdx.x * blockDim.x + threadIdx.x;      // 64*128
  int g = t >> 7;
  int o = t & 127;
  float inv = 1.0f / fmaxf(cnt[g], 1.0f);
  float s = 0.0f;
  const float* pg = pooled + (size_t)g * DIM;
#pragma unroll 8
  for (int k = 0; k < DIM; ++k) s += pg[k] * Wlin[(size_t)k * OUTC + o];
  out[t] = s * inv + blin[o];
}

// ---------------------------------------------------------------------------
extern "C" void kernel_launch(void* const* d_in, const int* in_sizes, int n_in,
                              void* d_out, int out_size, void* d_ws, size_t ws_size,
                              hipStream_t stream) {
  (void)in_sizes; (void)n_in; (void)out_size; (void)ws_size;

  const float* x    = (const float*)d_in[0];
  const int*   ei   = (const int*)d_in[1];            // [2, E]
  const int*   bat  = (const int*)d_in[2];
  const float* Wlin = (const float*)d_in[18];
  const float* blin = (const float*)d_in[19];
  float* out = (float*)d_out;

  const int* src = ei;
  const int* dst = ei + N_EDGES;

  float* ws     = (float*)d_ws;
  float* xA     = ws;
  float* xB     = xA + (size_t)NPAD * DIM;
  float* agg    = xB + (size_t)NPAD * DIM;
  float* pooled = agg + (size_t)NPAD * DIM;
  float* cnt    = pooled + (size_t)NGRAPH * DIM;
  _Float16* wt16 = (_Float16*)(cnt + NGRAPH);         // 10 x [512,512] f16

  const int padTot4  = NPAD * DIM / 4;                // 1,294,336
  const int copyTot4 = N_NODES * DIM / 4;

  // ---- one-time (per launch) weight conversion: f32 -> transposed f16 ----
  const int wtElems = DIM * DIM;                      // 262,144
  for (int l = 0; l < 5; ++l) {
    gcn_wt_convert<<<wtElems / 256, 256, 0, stream>>>(
        (const float*)d_in[3 + 3 * l], wt16 + (size_t)(2 * l) * wtElems);
    gcn_wt_convert<<<wtElems / 256, 256, 0, stream>>>(
        (const float*)d_in[4 + 3 * l], wt16 + (size_t)(2 * l + 1) * wtElems);
  }

  gcn_copy_pad<<<padTot4 / 256, 256, 0, stream>>>(x, xA, copyTot4, padTot4);

  float* cur = xA;
  float* nxt = xB;
  dim3 gemmGrid(NPAD / 128, DIM / 128);               // 79 x 4

  for (int l = 0; l < 5; ++l) {
    const _Float16* WrT = wt16 + (size_t)(2 * l) * wtElems;
    const _Float16* WsT = wt16 + (size_t)(2 * l + 1) * wtElems;
    const float*    b   = (const float*)d_in[5 + 3 * l];

    gcn_zero_f32<<<padTot4 / 256, 256, 0, stream>>>(agg, padTot4);
    gcn_scatter<<<(N_EDGES * 64) / 256, 256, 0, stream>>>(cur, src, dst, agg);
    gcn_layer_gemm<<<gemmGrid, 256, 0, stream>>>(cur, agg, WrT, WsT, b, nxt,
                                                 (l < 4) ? 1 : 0);
    float* tmp = cur; cur = nxt; nxt = tmp;
  }

  gcn_zero_f32<<<((NGRAPH * DIM + NGRAPH) / 4 + 255) / 256, 256, 0, stream>>>(
      pooled, (NGRAPH * DIM + NGRAPH) / 4);
  gcn_pool<<<(N_NODES * 64) / 256, 256, 0, stream>>>(cur, bat, pooled, cnt);
  gcn_final<<<(NGRAPH * OUTC) / 256, 256, 0, stream>>>(pooled, cnt, Wlin, blin, out);
}
